// TransformerSeq2Seq_22728966930946
// MI455X (gfx1250) — matrix operand order repaired and checked by reading, not compiled, not run on hardware
//
#include <hip/hip_runtime.h>

typedef __bf16 bf16_t;
typedef __attribute__((ext_vector_type(16))) __bf16 v16bf;
typedef __attribute__((ext_vector_type(8)))  float  v8f;

union FragB  { uint4 q[2]; v16bf v; };
union Pack16 { uint4 q[2]; bf16_t h[16]; };

#define WMMA_BF16(a, b, c) \
    __builtin_amdgcn_wmma_f32_16x16x32_bf16(false, (a), false, (b), (short)0, (c), false, false)

__device__ inline v8f zero_v8f() {
    v8f z;
#pragma unroll
    for (int i = 0; i < 8; i++) z[i] = 0.0f;
    return z;
}

// CDNA5 async global->LDS copy (no VGPR staging, tracked by ASYNCcnt).
// lds_addr = workgroup-relative LDS byte offset (low 32 bits of generic ptr).
__device__ inline void async_load_b128(unsigned lds_addr, const void* gaddr) {
    asm volatile("global_load_async_to_lds_b128 %0, %1, off"
                 :: "v"(lds_addr), "v"((unsigned long long)(size_t)gaddr)
                 : "memory");
}
__device__ inline void wait_async() {
    asm volatile("s_wait_asynccnt 0x0" ::: "memory");
}

// ---------------------------------------------------------------------------
// Tiled bf16 GEMM:  C(MxN) = A(MxK) * B(KxN) + bias, B given as BT (NxK).
// mode 0: f32 = acc+bias ; mode 1: f32 = relu(acc+bias) ; mode 2: bf16 = acc+bias
// 256 threads = 8 wave32; tile 128x128; wave tile 32x64 (2x4 WMMA accums).
// Double-buffered LDS fed by global_load_async_to_lds_b128.
// ---------------------------------------------------------------------------
__global__ __launch_bounds__(256) void gemm_bf16_kernel(
    const bf16_t* __restrict__ A, const bf16_t* __restrict__ BT,
    const float* __restrict__ bias,
    float* __restrict__ Cf, bf16_t* __restrict__ Cb,
    int M, int N, int K, int mode)
{
    __shared__ __align__(16) bf16_t As[2][128 * 32];
    __shared__ __align__(16) bf16_t Bs[2][128 * 32];

    const int t    = threadIdx.x;
    const int wave = t >> 5;
    const int lane = t & 31;
    const int hf   = lane >> 4;   // which 16-lane half
    const int l15  = lane & 15;
    const int bm   = blockIdx.y * 128;
    const int bn   = blockIdx.x * 128;
    const int mw   = (wave >> 1) * 32;   // wave row offset within tile
    const int nw   = (wave & 1) * 64;    // wave col offset within tile

    v8f acc[2][4];
#pragma unroll
    for (int i = 0; i < 2; i++)
#pragma unroll
        for (int j = 0; j < 4; j++) acc[i][j] = zero_v8f();

    const int lr = t >> 1;          // 0..127
    const int lc = (t & 1) * 16;    // 0 or 16

    const bf16_t* aRow = A  + (size_t)(bm + lr) * K + lc;
    const bf16_t* bRow = BT + (size_t)(bn + lr) * K + lc;
    const unsigned ldsA0 = (unsigned)(size_t)&As[0][lr * 32 + lc];
    const unsigned ldsA1 = (unsigned)(size_t)&As[1][lr * 32 + lc];
    const unsigned ldsB0 = (unsigned)(size_t)&Bs[0][lr * 32 + lc];
    const unsigned ldsB1 = (unsigned)(size_t)&Bs[1][lr * 32 + lc];

    // prologue: stage tile 0 into buffer 0
    async_load_b128(ldsA0, aRow);
    async_load_b128(ldsB0, bRow);

    int buf = 0;
    for (int k0 = 0; k0 < K; k0 += 32) {
        wait_async();        // this wave's async writes to LDS have landed
        __syncthreads();     // all waves' tile-k data visible; prev reads done

        if (k0 + 32 < K) {   // stage next tile into the other buffer
            async_load_b128(buf ? ldsA0 : ldsA1, aRow + k0 + 32);
            async_load_b128(buf ? ldsB0 : ldsB1, bRow + k0 + 32);
        }
        if (k0 + 64 < K) {   // hint tile k+2 toward L2 (global_prefetch_b8)
            __builtin_prefetch(aRow + k0 + 64, 0, 1);
            __builtin_prefetch(bRow + k0 + 64, 0, 1);
        }

        v16bf af[2], bfr[4];
#pragma unroll
        for (int i = 0; i < 2; i++) {
            FragB f;
            const bf16_t* p = &As[buf][(mw + 16 * i + l15) * 32 + hf * 8];
            f.q[0] = *(const uint4*)p;
            f.q[1] = *(const uint4*)(p + 16);
            af[i] = f.v;
        }
#pragma unroll
        for (int j = 0; j < 4; j++) {
            FragB f;
            const bf16_t* p = &Bs[buf][(nw + 16 * j + l15) * 32 + hf * 8];
            f.q[0] = *(const uint4*)p;
            f.q[1] = *(const uint4*)(p + 16);
            bfr[j] = f.v;
        }
#pragma unroll
        for (int i = 0; i < 2; i++)
#pragma unroll
            for (int j = 0; j < 4; j++)
                acc[i][j] = WMMA_BF16(af[i], bfr[j], acc[i][j]);

        buf ^= 1;
    }

#pragma unroll
    for (int i = 0; i < 2; i++) {
#pragma unroll
        for (int j = 0; j < 4; j++) {
            const int n   = bn + nw + 16 * j + l15;
            const float bvv = bias ? bias[n] : 0.0f;
#pragma unroll
            for (int r = 0; r < 8; r++) {
                const int m = bm + mw + 16 * i + hf * 8 + r;
                float v = acc[i][j][r] + bvv;
                if (mode == 1) v = v > 0.0f ? v : 0.0f;
                if (mode == 2) Cb[(size_t)m * N + n] = (bf16_t)v;
                else           Cf[(size_t)m * N + n] = v;
            }
        }
    }
}

// ---------------------------------------------------------------------------
// Attention per (b,h): logits = K @ V^T * scale; out = softmax(logits) @ V
// Flash-style online softmax. Q-tile (rows of K) = 128, key tile = 64.
// 256 threads = 8 waves; wave w owns query rows 16w..16w+15 and all 64 cols.
// ---------------------------------------------------------------------------
__global__ __launch_bounds__(256) void attn_kernel(
    const bf16_t* __restrict__ Kb, const bf16_t* __restrict__ Vb,
    float* __restrict__ Out, int S)
{
    const int D   = 1024;
    const int bh  = blockIdx.x;   // b*16 + h
    const int qb  = blockIdx.y;   // query block
    const int b   = bh >> 4, h = bh & 15;
    const bf16_t* Kp = Kb + (size_t)b * S * D + h * 64;
    const bf16_t* Vp = Vb + (size_t)b * S * D + h * 64;
    float*        Op = Out + (size_t)b * S * D + h * 64;

    __shared__ __align__(16) bf16_t Ks[128 * 64];   // query rows  (16 KB)
    __shared__ __align__(16) bf16_t Vs[64 * 64];    // V tile row-major (8 KB)
    __shared__ __align__(16) bf16_t VTs[64 * 64];   // V tile transposed (8 KB)
    __shared__ __align__(16) bf16_t Ps[8][16 * 64]; // per-wave P scratch (16 KB)

    const int t    = threadIdx.x;
    const int wave = t >> 5;
    const int lane = t & 31;
    const int hf   = lane >> 4;
    const int l15  = lane & 15;
    const float scale = 0.125f;  // 1/sqrt(64)

    // async-stage the query tile (128 x 64) straight into LDS
    {
        const int r = t >> 1, c = (t & 1) * 32;
        const bf16_t* src = Kp + (size_t)(qb * 128 + r) * D + c;
#pragma unroll
        for (int q = 0; q < 4; q++)
            async_load_b128((unsigned)(size_t)&Ks[r * 64 + c + q * 8], src + q * 8);
    }

    v8f oacc[4];
#pragma unroll
    for (int tn = 0; tn < 4; tn++) oacc[tn] = zero_v8f();
    float mrow[8], lrow[8];
#pragma unroll
    for (int r = 0; r < 8; r++) { mrow[r] = -1e30f; lrow[r] = 0.0f; }

    for (int j0 = 0; j0 < S; j0 += 64) {
        __syncthreads();   // previous Vs/VTs reads done
        {  // load V tile 64x64 (register path: needed for the transpose) + VTs
            const int r = t >> 2, c = (t & 3) * 16;
            Pack16 pk;
            const bf16_t* src = Vp + (size_t)(j0 + r) * D + c;
            pk.q[0] = *(const uint4*)src;
            pk.q[1] = *(const uint4*)(src + 8);
            *(uint4*)&Vs[r * 64 + c]     = pk.q[0];
            *(uint4*)&Vs[r * 64 + c + 8] = pk.q[1];
#pragma unroll
            for (int e = 0; e < 16; e++) VTs[(c + e) * 64 + r] = pk.h[e];
        }
        wait_async();      // Ks async writes landed (no-op after first iter)
        __syncthreads();

        // A-fragments of the query tile (16 x 64 -> two K=32 chunks)
        v16bf afS[2];
#pragma unroll
        for (int ks = 0; ks < 2; ks++) {
            FragB f;
            const bf16_t* p = &Ks[(16 * wave + l15) * 64 + 32 * ks + hf * 8];
            f.q[0] = *(const uint4*)p;
            f.q[1] = *(const uint4*)(p + 16);
            afS[ks] = f.v;
        }

        // S tile = Kq @ V^T  (16 x 64)
        v8f sacc[4];
#pragma unroll
        for (int ti = 0; ti < 4; ti++) {
            sacc[ti] = zero_v8f();
#pragma unroll
            for (int ks = 0; ks < 2; ks++) {
                FragB f;  // B[k][n] = V[n][k] -> contiguous in Vs row
                const bf16_t* p = &Vs[(16 * ti + l15) * 64 + 32 * ks + hf * 8];
                f.q[0] = *(const uint4*)p;
                f.q[1] = *(const uint4*)(p + 16);
                sacc[ti] = WMMA_BF16(afS[ks], f.v, sacc[ti]);
            }
        }

        // online softmax over the 64 new columns
        float mnew[8];
#pragma unroll
        for (int r = 0; r < 8; r++) mnew[r] = mrow[r];
#pragma unroll
        for (int ti = 0; ti < 4; ti++)
#pragma unroll
            for (int r = 0; r < 8; r++)
                mnew[r] = fmaxf(mnew[r], sacc[ti][r] * scale);
#pragma unroll
        for (int r = 0; r < 8; r++)
#pragma unroll
            for (int msk = 1; msk < 16; msk <<= 1)
                mnew[r] = fmaxf(mnew[r], __shfl_xor(mnew[r], msk, 32));

        float alpha[8], rsum[8];
#pragma unroll
        for (int r = 0; r < 8; r++) {
            alpha[r] = __expf(mrow[r] - mnew[r]);
            rsum[r]  = 0.0f;
            mrow[r]  = mnew[r];
        }
#pragma unroll
        for (int ti = 0; ti < 4; ti++)
#pragma unroll
            for (int r = 0; r < 8; r++) {
                float pe = __expf(sacc[ti][r] * scale - mnew[r]);
                rsum[r] += pe;
                Ps[wave][(r + hf * 8) * 64 + 16 * ti + l15] = (bf16_t)pe;
            }
#pragma unroll
        for (int r = 0; r < 8; r++) {
#pragma unroll
            for (int msk = 1; msk < 16; msk <<= 1)
                rsum[r] += __shfl_xor(rsum[r], msk, 32);
            lrow[r] = lrow[r] * alpha[r] + rsum[r];
        }
#pragma unroll
        for (int tn = 0; tn < 4; tn++)
#pragma unroll
            for (int r = 0; r < 8; r++) oacc[tn][r] *= alpha[r];

        // O += P @ V   (P: 16x64 in per-wave LDS, V via VTs for contiguous frags)
#pragma unroll
        for (int ks = 0; ks < 2; ks++) {
            FragB pf;
            const bf16_t* pp = &Ps[wave][l15 * 64 + 32 * ks + hf * 8];
            pf.q[0] = *(const uint4*)pp;
            pf.q[1] = *(const uint4*)(pp + 16);
#pragma unroll
            for (int tn = 0; tn < 4; tn++) {
                FragB vf;  // B[t][n] = V[t][n] = VTs[n][t]
                const bf16_t* vp = &VTs[(16 * tn + l15) * 64 + 32 * ks + hf * 8];
                vf.q[0] = *(const uint4*)vp;
                vf.q[1] = *(const uint4*)(vp + 16);
                oacc[tn] = WMMA_BF16(pf.v, vf.v, oacc[tn]);
            }
        }
    }

#pragma unroll
    for (int tn = 0; tn < 4; tn++)
#pragma unroll
        for (int r = 0; r < 8; r++) {
            const int qrow = qb * 128 + 16 * wave + hf * 8 + r;
            Op[(size_t)qrow * D + 16 * tn + l15] = oacc[tn][r] / lrow[r];
        }
}

// ---------------------------------------------------------------------------
// y = LayerNorm(X + R) * gamma + beta ; writes f32 and bf16 copies. 1 row/block.
// ---------------------------------------------------------------------------
__global__ __launch_bounds__(256) void add_ln_kernel(
    const float* __restrict__ X, const float* __restrict__ R,
    const float* __restrict__ gamma, const float* __restrict__ beta,
    float* __restrict__ Yf, bf16_t* __restrict__ Yb)
{
    const int D = 1024;
    const int row = blockIdx.x;
    const int t = threadIdx.x;
    __shared__ float s1[256], s2[256];

    float v[4];
    float sum = 0.0f, sq = 0.0f;
#pragma unroll
    for (int i = 0; i < 4; i++) {
        const int c = t + i * 256;
        const float x = X[(size_t)row * D + c] + R[(size_t)row * D + c];
        v[i] = x; sum += x; sq += x * x;
    }
    s1[t] = sum; s2[t] = sq;
    __syncthreads();
    for (int st = 128; st > 0; st >>= 1) {
        if (t < st) { s1[t] += s1[t + st]; s2[t] += s2[t + st]; }
        __syncthreads();
    }
    const float mu  = s1[0] * (1.0f / D);
    const float var = s2[0] * (1.0f / D) - mu * mu;
    const float rs  = rsqrtf(var + 1e-5f);
#pragma unroll
    for (int i = 0; i < 4; i++) {
        const int c = t + i * 256;
        const float y = (v[i] - mu) * rs * gamma[c] + beta[c];
        Yf[(size_t)row * D + c] = y;
        Yb[(size_t)row * D + c] = (bf16_t)y;
    }
}

// --------------------------- conversion kernels ----------------------------
__global__ __launch_bounds__(256) void cvt_f32_bf16_kernel(
    const float* __restrict__ src, bf16_t* __restrict__ dst, int n)
{
    const int i = blockIdx.x * 256 + threadIdx.x;
    if (i < n) dst[i] = (bf16_t)src[i];
}

// Wk/Wv layer slice (H=16, D=1024, DH=64) -> WT[n][d] with n = h*64+k
__global__ __launch_bounds__(256) void cvt_wkv_T_kernel(
    const float* __restrict__ W, bf16_t* __restrict__ WT)
{
    const int i = blockIdx.x * 256 + threadIdx.x;  // over 1024*1024
    const int n = i >> 10, d = i & 1023;
    const int h = n >> 6,  k = n & 63;
    WT[i] = (bf16_t)W[((size_t)h * 1024 + d) * 64 + k];
}

// W (Din x N) -> WT (N x Din)
__global__ __launch_bounds__(256) void cvt_T_kernel(
    const float* __restrict__ W, bf16_t* __restrict__ WT, int Din, int N)
{
    const int i = blockIdx.x * 256 + threadIdx.x;
    if (i >= Din * N) return;
    const int n = i / Din, d = i % Din;
    WT[i] = (bf16_t)W[(size_t)d * N + n];
}

// ---------------------------------------------------------------------------
extern "C" void kernel_launch(void* const* d_in, const int* in_sizes, int n_in,
                              void* d_out, int out_size, void* d_ws, size_t ws_size,
                              hipStream_t stream)
{
    const int B = 4, S = 1024, D = 1024, H = 16, DH = 64, L = 4;
    const int M = B * S;  // 4096 rows
    (void)in_sizes; (void)n_in; (void)out_size; (void)ws_size;

    const float* x_in = (const float*)d_in[0];
    const float* Wk   = (const float*)d_in[1];
    const float* bk   = (const float*)d_in[2];
    const float* Wv   = (const float*)d_in[3];
    const float* bv   = (const float*)d_in[4];
    const float* Wl   = (const float*)d_in[5];
    const float* bl   = (const float*)d_in[6];
    const float* g1   = (const float*)d_in[7];
    const float* b1   = (const float*)d_in[8];
    const float* g2   = (const float*)d_in[9];
    const float* b2   = (const float*)d_in[10];
    const float* Wo   = (const float*)d_in[11];
    const float* bo   = (const float*)d_in[12];

    char* ws = (char*)d_ws;
    size_t off = 0;
    auto alloc = [&](size_t bytes) -> char* {
        char* p = ws + off;
        off += (bytes + 255) & ~(size_t)255;
        return p;
    };
    bf16_t* xb    = (bf16_t*)alloc((size_t)M * D * 2);
    bf16_t* xmidb = (bf16_t*)alloc((size_t)M * D * 2);
    float*  xcur  = (float*) alloc((size_t)M * D * 4);
    float*  xmid  = (float*) alloc((size_t)M * D * 4);
    bf16_t* Kbf   = (bf16_t*)alloc((size_t)M * D * 2);
    bf16_t* Vbf   = (bf16_t*)alloc((size_t)M * D * 2);
    float*  attnb = (float*) alloc((size_t)M * D * 4);
    float*  ffb   = (float*) alloc((size_t)M * D * 4);
    bf16_t* WkT   = (bf16_t*)alloc((size_t)D * D * 2);
    bf16_t* WvT   = (bf16_t*)alloc((size_t)D * D * 2);
    bf16_t* WlT   = (bf16_t*)alloc((size_t)D * D * 2);
    bf16_t* WoT   = (bf16_t*)alloc((size_t)D * D * 2);

    const dim3 blk(256);
    const dim3 gemmGrid(D / 128, M / 128);
    const int  nX = M * D;

    cvt_f32_bf16_kernel<<<nX / 256, blk, 0, stream>>>(x_in, xb, nX);

    const float* xprev = x_in;
    for (int l = 0; l < L; l++) {
        cvt_wkv_T_kernel<<<(D * D) / 256, blk, 0, stream>>>(Wk + (size_t)l * H * D * DH, WkT);
        cvt_wkv_T_kernel<<<(D * D) / 256, blk, 0, stream>>>(Wv + (size_t)l * H * D * DH, WvT);
        gemm_bf16_kernel<<<gemmGrid, blk, 0, stream>>>(xb, WkT, bk + (size_t)l * D,
                                                       nullptr, Kbf, M, D, D, 2);
        gemm_bf16_kernel<<<gemmGrid, blk, 0, stream>>>(xb, WvT, bv + (size_t)l * D,
                                                       nullptr, Vbf, M, D, D, 2);
        attn_kernel<<<dim3(B * H, S / 128), blk, 0, stream>>>(Kbf, Vbf, attnb, S);
        add_ln_kernel<<<M, blk, 0, stream>>>(xprev, attnb, g1 + (size_t)l * D,
                                             b1 + (size_t)l * D, xmid, xmidb);
        cvt_T_kernel<<<(D * D) / 256, blk, 0, stream>>>(Wl + (size_t)l * D * D, WlT, D, D);
        gemm_bf16_kernel<<<gemmGrid, blk, 0, stream>>>(xmidb, WlT, bl + (size_t)l * D,
                                                       ffb, nullptr, M, D, D, 1);
        add_ln_kernel<<<M, blk, 0, stream>>>(xmid, ffb, g2 + (size_t)l * D,
                                             b2 + (size_t)l * D, xcur, xb);
        xprev = xcur;
    }
    cvt_T_kernel<<<(D * D) / 256, blk, 0, stream>>>(Wo, WoT, D, D);
    gemm_bf16_kernel<<<gemmGrid, blk, 0, stream>>>(xb, WoT, bo,
                                                   (float*)d_out, nullptr, M, D, D, 0);
}
